// TFtoTGShortcut_4801773437355
// MI455X (gfx1250) — matrix operand order repaired and checked by reading, not compiled, not run on hardware
//
#include <hip/hip_runtime.h>
#include <hip/hip_bf16.h>
#include <math.h>

// Problem constants (match the reference)
constexpr int G = 20000;
constexpr int T = 1500;
constexpr int D = 256;
constexpr int B = 512;
constexpr int TOPK = 32;
constexpr float THRESH     = 0.5f;
constexpr float PRIOR      = 0.5f;
constexpr float BIG_NEG    = -10000.0f;
constexpr float EPS        = 1e-8f;
constexpr float INV_SQRT_D = 0.0625f; // 1/sqrt(256)

// GEMM macro-tiling: block = 8 waves = 128(G) x 64(T), wave = 32x32, k-step 32
constexpr int BLK_G = 128;
constexpr int BLK_T = 64;
constexpr int KSTEP = 32;
constexpr int LDS_STRIDE = 40;   // 32 bf16 k-elements + 8 pad (80B rows, 16B aligned)

typedef __attribute__((ext_vector_type(16))) __bf16 v16bf;
typedef __attribute__((ext_vector_type(8)))  __bf16 v8bf;
typedef __attribute__((ext_vector_type(8)))  float  v8f;

// ---------------------------------------------------------------------------
// Kernel 1: sim = mask(tg_emb @ tf_id_emb.T / 16) via v_wmma_f32_16x16x32_bf16
// LDS-tiled: A(128x32) and B(64x32) staged as bf16 per k-step; each wave does
// a 32x32 quadrant (4 accumulators, 4 WMMAs per k-step). Masked/clipped
// logits are written into the attn region of d_out.
// ---------------------------------------------------------------------------
__global__ __launch_bounds__(256) void sim_wmma_kernel(
    const float* __restrict__ tg,      // [G, D]
    const float* __restrict__ tf,      // [T, D]
    const float* __restrict__ mmask,   // [G, T]
    float* __restrict__ sim)           // [G, T] (attn region of d_out)
{
    __shared__ __attribute__((aligned(16))) __bf16 ldsA[BLK_G * LDS_STRIDE]; // 10 KB
    __shared__ __attribute__((aligned(16))) __bf16 ldsB[BLK_T * LDS_STRIDE]; //  5 KB

    const int tid  = threadIdx.x;
    const int lane = tid & 31;
    const int wave = tid >> 5;

    const int gBase = blockIdx.x * BLK_G;
    const int tBase = blockIdx.y * BLK_T;

    // wave quadrant: 4 waves along G (32 rows each), 2 along T (32 cols each)
    const int waveG = (wave & 3) * 32;
    const int waveT = (wave >> 2) * 32;

    const int  m  = lane & 15;   // row within 16x16 tile (A/C)
    const int  n  = lane & 15;   // col within 16x16 tile (B/C)
    const bool hi = (lane >= 16);

    v8f acc00 = {}, acc01 = {}, acc10 = {}, acc11 = {};

    for (int kk = 0; kk < D; kk += KSTEP) {
        // ---- stage A: rows gBase..+128, k kk..+32 (bf16 in LDS) ----
        {
            const int r = tid >> 1;              // 0..127
            const int h = (tid & 1) * 16;        // k-half: 0 or 16
            int grow = gBase + r; if (grow >= G) grow = G - 1;
            const float4* src = (const float4*)(tg + (size_t)grow * D + kk + h);
            const float4 f0 = src[0], f1 = src[1], f2 = src[2], f3 = src[3];
            __bf16* dst = ldsA + r * LDS_STRIDE + h;
            dst[0]=(__bf16)f0.x; dst[1]=(__bf16)f0.y; dst[2]=(__bf16)f0.z; dst[3]=(__bf16)f0.w;
            dst[4]=(__bf16)f1.x; dst[5]=(__bf16)f1.y; dst[6]=(__bf16)f1.z; dst[7]=(__bf16)f1.w;
            dst[8]=(__bf16)f2.x; dst[9]=(__bf16)f2.y; dst[10]=(__bf16)f2.z; dst[11]=(__bf16)f2.w;
            dst[12]=(__bf16)f3.x; dst[13]=(__bf16)f3.y; dst[14]=(__bf16)f3.z; dst[15]=(__bf16)f3.w;
        }
        // ---- stage B: cols tBase..+64 (rows of tf_id_emb), k kk..+32 ----
        if (tid < 128) {
            const int r = tid >> 1;              // 0..63 (T column)
            const int h = (tid & 1) * 16;
            int col = tBase + r; if (col >= T) col = T - 1;  // clamped; discarded at store
            const float4* src = (const float4*)(tf + (size_t)col * D + kk + h);
            const float4 f0 = src[0], f1 = src[1], f2 = src[2], f3 = src[3];
            __bf16* dst = ldsB + r * LDS_STRIDE + h;
            dst[0]=(__bf16)f0.x; dst[1]=(__bf16)f0.y; dst[2]=(__bf16)f0.z; dst[3]=(__bf16)f0.w;
            dst[4]=(__bf16)f1.x; dst[5]=(__bf16)f1.y; dst[6]=(__bf16)f1.z; dst[7]=(__bf16)f1.w;
            dst[8]=(__bf16)f2.x; dst[9]=(__bf16)f2.y; dst[10]=(__bf16)f2.z; dst[11]=(__bf16)f2.w;
            dst[12]=(__bf16)f3.x; dst[13]=(__bf16)f3.y; dst[14]=(__bf16)f3.z; dst[15]=(__bf16)f3.w;
        }
        __syncthreads();

        // ---- fragments from LDS ----
        // A (16-bit 16x32 layout): lanes 0-15 K={0..7,16..23}, lanes 16-31 K={8..15,24..31}
        v16bf aF[2];
        #pragma unroll
        for (int i = 0; i < 2; ++i) {
            const __bf16* p = ldsA + (waveG + i * 16 + m) * LDS_STRIDE + (hi ? 8 : 0);
            const v8bf lo  = *(const v8bf*)p;
            const v8bf hi8 = *(const v8bf*)(p + 16);
            #pragma unroll
            for (int e = 0; e < 8; ++e) { aF[i][e] = lo[e]; aF[i][e + 8] = hi8[e]; }
        }
        // B (16-bit 32x16 layout): lanes 0-15 K={0..7,16..23}, lanes 16-31 K={8..15,24..31}
        v16bf bF[2];
        #pragma unroll
        for (int j = 0; j < 2; ++j) {
            const __bf16* p = ldsB + (waveT + j * 16 + n) * LDS_STRIDE + (hi ? 8 : 0);
            const v8bf c0 = *(const v8bf*)p;
            const v8bf c1 = *(const v8bf*)(p + 16);
            #pragma unroll
            for (int e = 0; e < 8; ++e) { bF[j][e] = c0[e]; bF[j][e + 8] = c1[e]; }
        }

        acc00 = __builtin_amdgcn_wmma_f32_16x16x32_bf16(false, aF[0], false, bF[0], (short)0, acc00, false, false);
        acc01 = __builtin_amdgcn_wmma_f32_16x16x32_bf16(false, aF[0], false, bF[1], (short)0, acc01, false, false);
        acc10 = __builtin_amdgcn_wmma_f32_16x16x32_bf16(false, aF[1], false, bF[0], (short)0, acc10, false, false);
        acc11 = __builtin_amdgcn_wmma_f32_16x16x32_bf16(false, aF[1], false, bF[1], (short)0, acc11, false, false);

        __syncthreads();
    }

    // ---- store with mask + prior + clip ----
    // C/D layout: VGPR r, lanes 0-15 -> (M=r, N=lane&15); lanes 16-31 -> (M=8+r)
    v8f* accs[2][2] = { { &acc00, &acc01 }, { &acc10, &acc11 } };
    #pragma unroll
    for (int i = 0; i < 2; ++i) {
        #pragma unroll
        for (int j = 0; j < 2; ++j) {
            const v8f a = *accs[i][j];
            const int col = tBase + waveT + j * 16 + n;
            if (col < T) {
                #pragma unroll
                for (int r = 0; r < 8; ++r) {
                    const int grow = gBase + waveG + i * 16 + (hi ? (r + 8) : r);
                    if (grow < G) {
                        const float mv = mmask[(size_t)grow * T + col];
                        float s = a[r] * INV_SQRT_D;
                        s = (mv >= THRESH) ? (s + PRIOR * mv) : BIG_NEG;
                        s = fminf(fmaxf(s, -50.0f), 50.0f);
                        sim[(size_t)grow * T + col] = s;
                    }
                }
            }
        }
    }
}

// ---------------------------------------------------------------------------
// Kernel 2: per-row masked softmax + top-32 sparsify + renormalize.
// One 256-thread block per gene row; in-place sim -> attn; topk pairs to ws.
// ---------------------------------------------------------------------------
__global__ __launch_bounds__(256) void softmax_topk_kernel(
    float* __restrict__ attn,          // [G, T] in: sim, out: attn
    int*   __restrict__ topkIdx,       // [G, TOPK]
    float* __restrict__ topkW)         // [G, TOPK]
{
    __shared__ float e[T + 4];
    __shared__ float red[256];
    __shared__ int   redi[256];
    __shared__ float selV[TOPK];
    __shared__ int   selI[TOPK];
    __shared__ float sDen;

    const int g   = blockIdx.x;
    const int tid = threadIdx.x;
    float* row = attn + (size_t)g * T;

    for (int i = tid; i < T; i += 256) e[i] = row[i];
    __syncthreads();

    // row max
    float m = -1e30f;
    for (int i = tid; i < T; i += 256) m = fmaxf(m, e[i]);
    red[tid] = m; __syncthreads();
    for (int s = 128; s > 0; s >>= 1) {
        if (tid < s) red[tid] = fmaxf(red[tid], red[tid + s]);
        __syncthreads();
    }
    const float rowMax = red[0];
    __syncthreads();

    // exp + sum
    float zloc = 0.0f;
    for (int i = tid; i < T; i += 256) {
        const float v = __expf(e[i] - rowMax);
        e[i] = v;
        zloc += v;
    }
    red[tid] = zloc; __syncthreads();
    for (int s = 128; s > 0; s >>= 1) {
        if (tid < s) red[tid] += red[tid + s];
        __syncthreads();
    }
    const float Z = red[0];
    __syncthreads();

    // iterative top-32 (ties -> lowest index, like lax.top_k)
    for (int k = 0; k < TOPK; ++k) {
        float bv = -1.0f; int bi = T;
        for (int i = tid; i < T; i += 256) {
            const float v = e[i];
            if (v > bv || (v == bv && i < bi)) { bv = v; bi = i; }
        }
        red[tid] = bv; redi[tid] = bi; __syncthreads();
        for (int s = 128; s > 0; s >>= 1) {
            if (tid < s) {
                const float ov = red[tid + s]; const int oi = redi[tid + s];
                if (ov > red[tid] || (ov == red[tid] && oi < redi[tid])) {
                    red[tid] = ov; redi[tid] = oi;
                }
            }
            __syncthreads();
        }
        if (tid == 0) {
            selV[k] = red[0];
            selI[k] = redi[0];
            e[redi[0]] = -1.0f;            // remove from candidate set (e >= 0)
        }
        __syncthreads();
    }

    // denom: (e/Z)/(Stopk/Z + EPS) = e/(Stopk + Z*EPS)
    if (tid == 0) {
        float s = 0.0f;
        #pragma unroll
        for (int k = 0; k < TOPK; ++k) s += selV[k];
        sDen = s + Z * EPS;
    }
    __syncthreads();
    const float inv = 1.0f / sDen;

    // write sparse attn row
    for (int i = tid; i < T; i += 256) row[i] = 0.0f;
    __syncthreads();
    if (tid < TOPK) {
        const float w = selV[tid] * inv;
        row[selI[tid]]          = w;
        topkIdx[g * TOPK + tid] = selI[tid];
        topkW[g * TOPK + tid]   = w;
    }
}

// ---------------------------------------------------------------------------
// Kernel 3: output[b,g] = scale * sum_j w[g,j] * tf_expr[b, idx[g,j]]
// Block = (256 genes, one cell b); tf_expr row in LDS; coalesced stores.
// ---------------------------------------------------------------------------
__global__ __launch_bounds__(256) void combine_kernel(
    const float* __restrict__ tfExpr,   // [B, T]
    const int*   __restrict__ topkIdx,  // [G, TOPK]
    const float* __restrict__ topkW,    // [G, TOPK]
    const float* __restrict__ scaleP,   // [1]
    float* __restrict__ out)            // [B, G]
{
    __shared__ float erow[T + 4];
    const int b = blockIdx.y;
    const int g = blockIdx.x * 256 + threadIdx.x;

    for (int i = threadIdx.x; i < T; i += 256) erow[i] = tfExpr[(size_t)b * T + i];
    __syncthreads();

    if (g < G) {
        const int*   idx = topkIdx + g * TOPK;
        const float* w   = topkW   + g * TOPK;
        float acc = 0.0f;
        #pragma unroll
        for (int j = 0; j < TOPK; ++j) acc += w[j] * erow[idx[j]];
        out[(size_t)b * G + g] = scaleP[0] * acc;
    }
}

// ---------------------------------------------------------------------------
extern "C" void kernel_launch(void* const* d_in, const int* in_sizes, int n_in,
                              void* d_out, int out_size, void* d_ws, size_t ws_size,
                              hipStream_t stream) {
    const float* tg     = (const float*)d_in[0];   // [G, D]
    const float* tf     = (const float*)d_in[1];   // [T, D]
    const float* tfExpr = (const float*)d_in[2];   // [B, T]
    const float* mmask  = (const float*)d_in[3];   // [G, T]
    const float* scale  = (const float*)d_in[4];   // [1]

    float* out  = (float*)d_out;                   // [B, G]
    float* attn = out + (size_t)B * G;             // [G, T]

    int*   topkIdx = (int*)d_ws;                                             // 2.56 MB
    float* topkW   = (float*)((char*)d_ws + (size_t)G * TOPK * sizeof(int)); // 2.56 MB

    // 1) LDS-tiled WMMA GEMM + mask/prior/clip -> masked logits in attn region
    dim3 gridA((G + BLK_G - 1) / BLK_G, (T + BLK_T - 1) / BLK_T);  // (157, 24)
    sim_wmma_kernel<<<gridA, 256, 0, stream>>>(tg, tf, mmask, attn);

    // 2) per-row softmax + top-32 + renormalize (in place) + topk to ws
    softmax_topk_kernel<<<G, 256, 0, stream>>>(attn, topkIdx, topkW);

    // 3) sparse combine -> output
    dim3 gridC((G + 255) / 256, B);
    combine_kernel<<<gridC, 256, 0, stream>>>(tfExpr, topkIdx, topkW, scale, out);
}
